// GraphAutoencoder_80942953660708
// MI455X (gfx1250) — compile-verified, matrix operand order checked by reading, not run on hardware
//
#include <hip/hip_runtime.h>
#include <hip/hip_bf16.h>

typedef __bf16 bf16;
typedef __attribute__((ext_vector_type(16))) __bf16 v16bf;
typedef __attribute__((ext_vector_type(8)))  float  v8f;

#define N_NODES  50000
#define N_EDGES  800000
#define N_GRAPHS 512
#define IN_DIM   128
#define HID_DIM  128
#define LAT_DIM  64

__device__ __forceinline__ bf16 f2bf(float f) {
    union { float f; unsigned u; } c; c.f = f;
    unsigned r = c.u + 0x7FFFu + ((c.u >> 16) & 1u);   // round-to-nearest-even
    unsigned short h = (unsigned short)(r >> 16);
    union { unsigned short u; bf16 b; } o; o.u = h;
    return o.b;
}

// ---------------- elementwise utility kernels ----------------
__global__ void k_fill(float* p, float v, int n) {
    int i = blockIdx.x * blockDim.x + threadIdx.x;
    if (i < n) p[i] = v;
}

__global__ void k_rsqrt_inplace(float* p, int n) {
    int i = blockIdx.x * blockDim.x + threadIdx.x;
    if (i < n) p[i] = rsqrtf(p[i]);
}

__global__ void k_deg_accum(const int* __restrict__ dst, float* __restrict__ deg, int e) {
    int i = blockIdx.x * blockDim.x + threadIdx.x;
    if (i < e) atomicAdd(&deg[dst[i]], 1.0f);
}

__global__ void k_f32_to_bf16(const float* __restrict__ in, bf16* __restrict__ out, int n) {
    int i = blockIdx.x * blockDim.x + threadIdx.x;
    if (i < n) out[i] = f2bf(in[i]);
}

// W is (Fin, Fout) row-major; produce Wt (Fout, Fin) row-major in bf16
__global__ void k_weight_bf16_t(const float* __restrict__ W, bf16* __restrict__ Wt,
                                int Fin, int Fout) {
    int i = blockIdx.x * blockDim.x + threadIdx.x;
    if (i >= Fin * Fout) return;
    int k = i / Fout, n = i % Fout;
    Wt[n * Fin + k] = f2bf(W[i]);
}

// ---------------- WMMA bf16 GEMM: C[M,N] = A[M,K] * Bt[N,K]^T ----------------
// grid.x = M/16, blockDim.x = (N/16)*32 ; M multiple of 16, K multiple of 32.
__global__ __launch_bounds__(256)
void k_gemm_wmma_bf16(const bf16* __restrict__ A, const bf16* __restrict__ Bt,
                      float* __restrict__ C, int K, int N) {
    const int lane = threadIdx.x & 31;
    const int wave = threadIdx.x >> 5;
    const int row0 = blockIdx.x * 16;
    const int col0 = wave * 16;
    const int r    = lane & 15;
    const int kbp  = (lane >> 4) * 4;           // K-pair offset for hi half-wave

    const unsigned* a32 = (const unsigned*)(A  + (size_t)(row0 + r) * K);
    const unsigned* b32 = (const unsigned*)(Bt + (size_t)(col0 + r) * K);

    v8f c = {};
    for (int kk = 0; kk < K; kk += 32) {
        const int base = kk >> 1;               // pair index of chunk start
        union { v16bf v; unsigned u[8]; } a, b;
#pragma unroll
        for (int j = 0; j < 4; ++j) {
            a.u[j]     = a32[base + kbp + j];
            a.u[j + 4] = a32[base + 8 + kbp + j];
            b.u[j]     = b32[base + kbp + j];
            b.u[j + 4] = b32[base + 8 + kbp + j];
        }
        c = __builtin_amdgcn_wmma_f32_16x16x32_bf16(false, a.v, false, b.v,
                                                    (short)0, c, false, false);
    }

    // C/D layout: VGPR i, lanes 0-15 -> row i, lanes 16-31 -> row i+8; col = lane&15
    const int moff = (lane < 16) ? 0 : 8;
    float* out = C + (size_t)row0 * N + col0 + r;
#pragma unroll
    for (int i = 0; i < 8; ++i)
        out[(size_t)(i + moff) * N] = c[i];
}

// ---------------- edge gather/scale/scatter-add ----------------
// one thread per (edge, 4 features)
__global__ void k_edge_scatter(const float* __restrict__ h,
                               const int* __restrict__ src, const int* __restrict__ dst,
                               const float* __restrict__ dis,
                               float* __restrict__ out, int E, int F) {
    long tid = (long)blockIdx.x * blockDim.x + threadIdx.x;
    int per = F >> 2;
    long e = tid / per;
    if (e >= E) return;
    int f4 = (int)(tid % per) << 2;
    int s = src[e], d = dst[e];
    float nrm = dis[s] * dis[d];
    const float4 hv = *(const float4*)(h + (size_t)s * F + f4);
    float* o = out + (size_t)d * F + f4;
    atomicAdd(o + 0, hv.x * nrm);
    atomicAdd(o + 1, hv.y * nrm);
    atomicAdd(o + 2, hv.z * nrm);
    atomicAdd(o + 3, hv.w * nrm);
}

// out = agg + h*dis^2 + b, optional ReLU
__global__ void k_finalize(const float* __restrict__ agg, const float* __restrict__ h,
                           const float* __restrict__ dis, const float* __restrict__ bias,
                           float* __restrict__ out, int n, int F, int relu) {
    int i = blockIdx.x * blockDim.x + threadIdx.x;
    if (i >= n * F) return;
    int node = i / F, f = i % F;
    float d = dis[node];
    float v = agg[i] + h[i] * d * d + bias[f];
    out[i] = relu ? fmaxf(v, 0.0f) : v;
}

// ---------------- pooling ----------------
__global__ void k_pool_scatter(const float* __restrict__ z, const int* __restrict__ batch,
                               float* __restrict__ agg, int n) {
    int i = blockIdx.x * blockDim.x + threadIdx.x;
    if (i >= n * LAT_DIM) return;
    int node = i / LAT_DIM, f = i % LAT_DIM;
    atomicAdd(&agg[batch[node] * LAT_DIM + f], z[i]);
}

__global__ void k_pool_count(const int* __restrict__ batch, float* __restrict__ counts, int n) {
    int i = blockIdx.x * blockDim.x + threadIdx.x;
    if (i < n) atomicAdd(&counts[batch[i]], 1.0f);
}

__global__ void k_pool_div(float* __restrict__ agg, const float* __restrict__ counts) {
    int i = blockIdx.x * blockDim.x + threadIdx.x;
    if (i >= N_GRAPHS * LAT_DIM) return;
    agg[i] /= fmaxf(counts[i / LAT_DIM], 1.0f);
}

// ---------------- host orchestration ----------------
static inline size_t alup(size_t x) { return (x + 255) & ~(size_t)255; }

extern "C" void kernel_launch(void* const* d_in, const int* in_sizes, int n_in,
                              void* d_out, int out_size, void* d_ws, size_t ws_size,
                              hipStream_t stream) {
    (void)in_sizes; (void)n_in; (void)out_size; (void)ws_size;
    const float* x   = (const float*)d_in[0];
    const int*   ei  = (const int*)d_in[1];
    const int*   src = ei;
    const int*   dst = ei + N_EDGES;
    const int*   bat = (const int*)d_in[2];
    const float* W1 = (const float*)d_in[3];  const float* b1 = (const float*)d_in[4];
    const float* W2 = (const float*)d_in[5];  const float* b2 = (const float*)d_in[6];
    const float* W3 = (const float*)d_in[7];  const float* b3 = (const float*)d_in[8];
    const float* W4 = (const float*)d_in[9];  const float* b4 = (const float*)d_in[10];

    float* x_hat = (float*)d_out;                               // [N,128]
    float* z     = x_hat + (size_t)N_NODES * IN_DIM;            // [N,64]
    float* agg   = z + (size_t)N_NODES * LAT_DIM;               // [512,64]

    char* p = (char*)d_ws;
    float* dis    = (float*)p; p += alup((size_t)N_NODES * 4);
    float* counts = (float*)p; p += alup((size_t)N_GRAPHS * 4);
    bf16*  Wt1    = (bf16*)p;  p += alup((size_t)IN_DIM * HID_DIM * 2);
    bf16*  Wt2    = (bf16*)p;  p += alup((size_t)HID_DIM * LAT_DIM * 2);
    bf16*  Wt3    = (bf16*)p;  p += alup((size_t)LAT_DIM * HID_DIM * 2);
    bf16*  Wt4    = (bf16*)p;  p += alup((size_t)HID_DIM * IN_DIM * 2);
    bf16*  actb   = (bf16*)p;  p += alup((size_t)N_NODES * 128 * 2);
    float* hbuf   = (float*)p; p += alup((size_t)N_NODES * 128 * 4);
    float* aggb   = (float*)p; p += alup((size_t)N_NODES * 128 * 4);

    const int T = 256;
    auto blk = [](long n, int t) { return (unsigned)((n + t - 1) / t); };

    // ---- degrees -> dis = rsqrt(deg) ----
    k_fill<<<blk(N_NODES, T), T, 0, stream>>>(dis, 1.0f, N_NODES);
    k_deg_accum<<<blk(N_EDGES, T), T, 0, stream>>>(dst, dis, N_EDGES);
    k_rsqrt_inplace<<<blk(N_NODES, T), T, 0, stream>>>(dis, N_NODES);

    // ---- weights -> transposed bf16 ----
    k_weight_bf16_t<<<blk(IN_DIM * HID_DIM, T), T, 0, stream>>>(W1, Wt1, IN_DIM, HID_DIM);
    k_weight_bf16_t<<<blk(HID_DIM * LAT_DIM, T), T, 0, stream>>>(W2, Wt2, HID_DIM, LAT_DIM);
    k_weight_bf16_t<<<blk(LAT_DIM * HID_DIM, T), T, 0, stream>>>(W3, Wt3, LAT_DIM, HID_DIM);
    k_weight_bf16_t<<<blk(HID_DIM * IN_DIM, T), T, 0, stream>>>(W4, Wt4, HID_DIM, IN_DIM);

    // one GCN conv layer
    auto conv = [&](const float* in, int Fin, const bf16* Wt, int Fout,
                    const float* bias, float* out, int relu) {
        long nin = (long)N_NODES * Fin;
        long nout = (long)N_NODES * Fout;
        k_f32_to_bf16<<<blk(nin, T), T, 0, stream>>>(in, actb, (int)nin);
        k_gemm_wmma_bf16<<<N_NODES / 16, (Fout / 16) * 32, 0, stream>>>(actb, Wt, hbuf, Fin, Fout);
        k_fill<<<blk(nout, T), T, 0, stream>>>(aggb, 0.0f, (int)nout);
        long nscat = (long)N_EDGES * (Fout / 4);
        k_edge_scatter<<<blk(nscat, T), T, 0, stream>>>(hbuf, src, dst, dis, aggb, N_EDGES, Fout);
        k_finalize<<<blk(nout, T), T, 0, stream>>>(aggb, hbuf, dis, bias, out, N_NODES, Fout, relu);
    };

    // encoder: h1 = relu(conv(x, W1)) stored temporarily in x_hat region
    conv(x, IN_DIM, Wt1, HID_DIM, b1, x_hat, 1);
    // z = conv(h1, W2)
    conv(x_hat, HID_DIM, Wt2, LAT_DIM, b2, z, 0);

    // scatter-mean pooling
    k_fill<<<blk(N_GRAPHS * LAT_DIM, T), T, 0, stream>>>(agg, 0.0f, N_GRAPHS * LAT_DIM);
    k_fill<<<blk(N_GRAPHS, T), T, 0, stream>>>(counts, 0.0f, N_GRAPHS);
    k_pool_scatter<<<blk((long)N_NODES * LAT_DIM, T), T, 0, stream>>>(z, bat, agg, N_NODES);
    k_pool_count<<<blk(N_NODES, T), T, 0, stream>>>(bat, counts, N_NODES);
    k_pool_div<<<blk(N_GRAPHS * LAT_DIM, T), T, 0, stream>>>(agg, counts);

    // decoder: h2 = relu(conv(z, W3)) reuses x_hat region (h1 dead)
    conv(z, LAT_DIM, Wt3, HID_DIM, b3, x_hat, 1);
    // x_hat = conv(h2, W4); h2 is consumed by the bf16 convert before finalize overwrites
    conv(x_hat, HID_DIM, Wt4, IN_DIM, b4, x_hat, 0);
}
